// TransformerBlock_53644141527455
// MI455X (gfx1250) — compile-verified, matrix operand order checked by reading, not run on hardware
//
#include <hip/hip_runtime.h>
#include <hip/hip_bf16.h>
#include <math.h>

typedef __bf16 bf16;
typedef __attribute__((ext_vector_type(8)))  __bf16 v8bf;
typedef __attribute__((ext_vector_type(16))) __bf16 v16bf;
typedef __attribute__((ext_vector_type(8)))  float  v8f;
typedef int v4i __attribute__((vector_size(16)));
typedef __attribute__((address_space(1))) v4i* gptr_v4i;
typedef __attribute__((address_space(3))) v4i* lptr_v4i;

#define DEV __device__ __forceinline__

#if __has_builtin(__builtin_amdgcn_global_load_async_to_lds_b128) && \
    __has_builtin(__builtin_amdgcn_s_wait_asynccnt)
#define HAS_ASYNC_LDS 1
#else
#define HAS_ASYNC_LDS 0
#endif

static constexpr int Bsz = 2, Tt = 2048, Cc = 1024, Hh = 16, DKv = 64, HID = 8192;
static constexpr int Mrows = Bsz * Tt;  // 4096

DEV v8f wmma_bf(v16bf a, v16bf b, v8f c) {
  return __builtin_amdgcn_wmma_f32_16x16x32_bf16(false, a, false, b, (short)0, c,
                                                 false, false);
}

// Load one 16(M or N) x 32(K) bf16 fragment from a row-major buffer whose row
// index is the M/N coordinate and column index is K. Per CDNA5 ISA layout:
// lanes 0-15 -> row = row0+lane, halves 0..7 = K[grp*8 .. grp*8+7],
// halves 8..15 = K[16+grp*8 .. 16+grp*8+7], grp = lane/16.  Both 8-half groups
// are contiguous 16-byte loads.
DEV v16bf load_frag(const bf16* __restrict__ base, int row0, int ld, int k0, int lane) {
  int m   = row0 + (lane & 15);
  int grp = (lane >> 4) & 1;
  const bf16* p = base + (size_t)m * ld + k0 + grp * 8;
  v8bf lo = *reinterpret_cast<const v8bf*>(p);
  v8bf hi = *reinterpret_cast<const v8bf*>(p + 16);
  v16bf r;
#pragma unroll
  for (int i = 0; i < 8; ++i) { r[i] = lo[i]; r[8 + i] = hi[i]; }
  return r;
}

// 16-byte global -> LDS copy; async CDNA5 path (ASYNCcnt), sync fallback.
DEV void copy16_g2l(const bf16* __restrict__ g, bf16* l) {
#if HAS_ASYNC_LDS
  __builtin_amdgcn_global_load_async_to_lds_b128((gptr_v4i)g, (lptr_v4i)l, 0, 0);
#else
  *reinterpret_cast<v8bf*>(l) = *reinterpret_cast<const v8bf*>(g);
#endif
}
DEV void wait_async_lds() {
#if HAS_ASYNC_LDS
  __builtin_amdgcn_s_wait_asynccnt(0);
#endif
}

DEV float gelu_exact(float x) { return 0.5f * x * (1.0f + erff(x * 0.7071067811865476f)); }

DEV float wred_maxf(float v) {
#pragma unroll
  for (int o = 16; o > 0; o >>= 1) v = fmaxf(v, __shfl_xor(v, o, 32));
  return v;
}
DEV float wred_sumf(float v) {
#pragma unroll
  for (int o = 16; o > 0; o >>= 1) v += __shfl_xor(v, o, 32);
  return v;
}
DEV int wred_sumi(int v) {
#pragma unroll
  for (int o = 16; o > 0; o >>= 1) v += __shfl_xor(v, o, 32);
  return v;
}
DEV int lane_excl_scan(int v, int lane) {
  int x = v;
#pragma unroll
  for (int o = 1; o < 32; o <<= 1) { int y = __shfl_up(x, o, 32); if (lane >= o) x += y; }
  return x - v;
}
// monotonic float -> uint map (order preserving, incl. negatives)
DEV unsigned fmap(float f) {
  unsigned u = __float_as_uint(f);
  return (u & 0x80000000u) ? ~u : (u | 0x80000000u);
}

// ---------------------------------------------------------------- elementwise
__global__ void cvt_transpose_kernel(const float* __restrict__ W, bf16* __restrict__ Wt,
                                     int K, int N) {
  size_t idx = (size_t)blockIdx.x * blockDim.x + threadIdx.x;
  size_t tot = (size_t)K * N;
  if (idx >= tot) return;
  int k = (int)(idx / N), n = (int)(idx % N);
  Wt[(size_t)n * K + k] = (bf16)W[idx];
}

__global__ __launch_bounds__(256) void layernorm_kernel(const float* __restrict__ x,
                                                        const float* __restrict__ g,
                                                        const float* __restrict__ b,
                                                        bf16* __restrict__ out) {
  __shared__ float sred[256];
  int row = blockIdx.x, tid = threadIdx.x;
  const float* xr = x + (size_t)row * Cc;
  float s = 0.f;
  for (int i = tid; i < Cc; i += 256) s += xr[i];
  sred[tid] = s; __syncthreads();
  for (int st = 128; st > 0; st >>= 1) { if (tid < st) sred[tid] += sred[tid + st]; __syncthreads(); }
  float mean = sred[0] / (float)Cc; __syncthreads();
  float v = 0.f;
  for (int i = tid; i < Cc; i += 256) { float d = xr[i] - mean; v += d * d; }
  sred[tid] = v; __syncthreads();
  for (int st = 128; st > 0; st >>= 1) { if (tid < st) sred[tid] += sred[tid + st]; __syncthreads(); }
  float rstd = rsqrtf(sred[0] / (float)Cc + 1e-5f);
  for (int i = tid; i < Cc; i += 256)
    out[(size_t)row * Cc + i] = (bf16)((xr[i] - mean) * rstd * g[i] + b[i]);
}

__global__ void resid_kernel(const float* __restrict__ a, const float* __restrict__ y,
                             const float* __restrict__ alpha, float* __restrict__ out, int n) {
  int i = blockIdx.x * blockDim.x + threadIdx.x;
  if (i < n) out[i] = a[i] + alpha[0] * y[i];
}

__global__ void split_qkv_kernel(const bf16* __restrict__ qkv, bf16* __restrict__ Q,
                                 bf16* __restrict__ K, bf16* __restrict__ Vt) {
  size_t idx = (size_t)blockIdx.x * blockDim.x + threadIdx.x;
  if (idx >= (size_t)Mrows * 3 * Cc) return;
  int col = (int)(idx % (3 * Cc));
  int rw  = (int)(idx / (3 * Cc));
  int b = rw / Tt, t = rw % Tt;
  int part = col / Cc, c = col % Cc, h = c / DKv, d = c % DKv;
  float v = (float)qkv[idx];
  size_t bh = (size_t)b * Hh + h;
  if (part == 0)      Q[(bh * Tt + t) * DKv + d] = (bf16)(v * 0.125f); // 1/sqrt(64)
  else if (part == 1) K[(bh * Tt + t) * DKv + d] = (bf16)v;
  else                Vt[(bh * DKv + d) * Tt + t] = (bf16)v;
}

// ---------------------------------------------------------------- WMMA GEMM
// C[M,N] = A[M,K](bf16) * Bt[N,K]^T (bf16), f32 accumulate.
// Block tile 128x128, 8 waves in 4(M)x2(N), each wave 32x64 (2x4 wmma tiles).
// Double-buffered LDS staged with GLOBAL_LOAD_ASYNC_TO_LDS_B128 when available.
template <bool OUT_BF16, bool GELU>
__global__ __launch_bounds__(256) void gemm_bf16_kernel(
    const bf16* __restrict__ A, const bf16* __restrict__ Bt, void* __restrict__ Cout,
    const float* __restrict__ bias, int M, int N, int K) {
  __shared__ __align__(16) bf16 As[2][128 * 32];
  __shared__ __align__(16) bf16 Bs[2][128 * 32];
  int tid = threadIdx.x, lane = tid & 31, wave = tid >> 5;
  int wm = wave & 3, wn = wave >> 2;
  int bm = blockIdx.y * 128, bn = blockIdx.x * 128;

  auto stage = [&](int buf, int k0) {
#pragma unroll
    for (int i = 0; i < 2; ++i) {
      int c = tid + i * 256;                 // 512 chunks of 8 halves per tile
      int r = c >> 2, cc = (c & 3) * 8;
      copy16_g2l(&A[(size_t)(bm + r) * K + k0 + cc], &As[buf][r * 32 + cc]);
      copy16_g2l(&Bt[(size_t)(bn + r) * K + k0 + cc], &Bs[buf][r * 32 + cc]);
    }
  };

  v8f acc[2][4] = {};
  int nk = K / 32;
  stage(0, 0);
  wait_async_lds();
  __syncthreads();
  for (int kt = 0; kt < nk; ++kt) {
    int buf = kt & 1;
    if (kt + 1 < nk) stage(buf ^ 1, (kt + 1) * 32);   // overlap with WMMA below
    v16bf bf_[4];
#pragma unroll
    for (int j = 0; j < 4; ++j) bf_[j] = load_frag(&Bs[buf][0], wn * 64 + j * 16, 32, 0, lane);
#pragma unroll
    for (int i = 0; i < 2; ++i) {
      v16bf af = load_frag(&As[buf][0], wm * 32 + i * 16, 32, 0, lane);
#pragma unroll
      for (int j = 0; j < 4; ++j) acc[i][j] = wmma_bf(af, bf_[j], acc[i][j]);
    }
    wait_async_lds();
    __syncthreads();
  }
  int grp = lane >> 4, nn = lane & 15;
#pragma unroll
  for (int i = 0; i < 2; ++i)
#pragma unroll
    for (int j = 0; j < 4; ++j)
#pragma unroll
      for (int r = 0; r < 8; ++r) {
        int row = bm + wm * 32 + i * 16 + r + 8 * grp;
        int col = bn + wn * 64 + j * 16 + nn;
        float v = acc[i][j][r];
        if (bias) v += bias[col];
        if (GELU) v = gelu_exact(v);
        if (OUT_BF16) ((bf16*)Cout)[(size_t)row * N + col] = (bf16)v;
        else          ((float*)Cout)[(size_t)row * N + col] = v;
      }
}

// ---------------------------------------------------------------- attention
// One workgroup handles (b, h, 16 query rows): S = QK^T (wmma) -> dense softmax
// + exact top-64 sparse softmax (binary search on order-mapped bits, index-
// ordered tie break) -> gated blend into bf16 P -> O = P V (wmma).
__global__ __launch_bounds__(256) void attn_kernel(
    const bf16* __restrict__ Q, const bf16* __restrict__ Kmat,
    const bf16* __restrict__ Vt, const float* __restrict__ alpha_head,
    bf16* __restrict__ Ocat) {
  extern __shared__ char smem[];
  float* S = (float*)smem;                                  // 16*2048 f32
  bf16*  P = (bf16*)(smem + 16 * Tt * 4);                   // 16*2048 bf16
  float* red = (float*)(smem + 16 * Tt * 4 + 16 * Tt * 2);  // 4*16*16 f32

  int qt = blockIdx.x, h = blockIdx.y, b = blockIdx.z;
  size_t bh = (size_t)b * Hh + h;
  const bf16* Qb = Q    + bh * Tt * DKv;
  const bf16* Kb = Kmat + bh * Tt * DKv;
  const bf16* Vb = Vt   + bh * DKv * Tt;
  int q0 = qt * 16;
  int tid = threadIdx.x, lane = tid & 31, wave = tid >> 5;
  int grp = lane >> 4, nn = lane & 15;

  // ---- phase 1: scores (scale already folded into Q)
  v16bf qa0 = load_frag(Qb, q0, DKv, 0, lane);
  v16bf qa1 = load_frag(Qb, q0, DKv, 32, lane);
  for (int kb = wave; kb < Tt / 16; kb += 8) {
    v8f acc = {};
    acc = wmma_bf(qa0, load_frag(Kb, kb * 16, DKv, 0, lane), acc);
    acc = wmma_bf(qa1, load_frag(Kb, kb * 16, DKv, 32, lane), acc);
#pragma unroll
    for (int r = 0; r < 8; ++r) S[(r + 8 * grp) * Tt + kb * 16 + nn] = acc[r];
  }
  __syncthreads();

  // ---- phase 2: dense softmax + top-64 sparse softmax + gated blend
  float gate = 1.f / (1.f + __expf(-alpha_head[h]));
#pragma unroll 1
  for (int rr = 0; rr < 2; ++rr) {
    int row = wave * 2 + rr;
    const float* Srow = S + (size_t)row * Tt;
    int base = lane * 64;  // contiguous chunk keeps index order for tie-break
    float mx = -3.4e38f;
    for (int i = 0; i < 64; ++i) mx = fmaxf(mx, Srow[base + i]);
    mx = wred_maxf(mx);
    float dsum = 0.f;
    for (int i = 0; i < 64; ++i) dsum += __expf(Srow[base + i] - mx);
    dsum = wred_sumf(dsum);
    // exact 64th-largest via 32-bit binary search on monotone-mapped values
    unsigned thr = 0;
    for (int bit = 31; bit >= 0; --bit) {
      unsigned cand = thr | (1u << bit);
      int cnt = 0;
      for (int i = 0; i < 64; ++i) cnt += (fmap(Srow[base + i]) >= cand);
      if (wred_sumi(cnt) >= 64) thr = cand;
    }
    int cgt = 0, ceq = 0;
    for (int i = 0; i < 64; ++i) {
      unsigned u = fmap(Srow[base + i]);
      cgt += (u > thr); ceq += (u == thr);
    }
    int rkeep = 64 - wred_sumi(cgt);          // # of ==thr entries to keep
    int eq0 = lane_excl_scan(ceq, lane);      // rank of my first ==thr entry
    float ssum = 0.f;
    { int er = eq0;
      for (int i = 0; i < 64; ++i) {
        unsigned u = fmap(Srow[base + i]);
        bool keep;
        if (u == thr) { keep = (er < rkeep); ++er; } else keep = (u > thr);
        if (keep) ssum += __expf(Srow[base + i] - mx);
      } }
    ssum = wred_sumf(ssum);
    float dinv = 1.f / dsum, sinv = 1.f / ssum;
    { int er = eq0;
      for (int i = 0; i < 64; ++i) {
        float e = __expf(Srow[base + i] - mx);
        unsigned u = fmap(Srow[base + i]);
        bool keep;
        if (u == thr) { keep = (er < rkeep); ++er; } else keep = (u > thr);
        float p = gate * e * dinv + (1.f - gate) * (keep ? e * sinv : 0.f);
        P[(size_t)row * Tt + base + i] = (bf16)p;
      } }
  }
  __syncthreads();

  // ---- phase 3: O = P * V  (4 n-tiles x 2 K-halves across 8 waves)
  int ntile = wave & 3, half = wave >> 2;
  v8f acc = {};
  for (int ks = half * 32; ks < half * 32 + 32; ++ks) {
    v16bf af = load_frag(P, 0, Tt, ks * 32, lane);
    v16bf bf_ = load_frag(Vb, ntile * 16, Tt, ks * 32, lane);
    acc = wmma_bf(af, bf_, acc);
  }
  if (half == 1) {
#pragma unroll
    for (int r = 0; r < 8; ++r) red[ntile * 256 + (r + 8 * grp) * 16 + nn] = acc[r];
  }
  __syncthreads();
  if (half == 0) {
#pragma unroll
    for (int r = 0; r < 8; ++r) {
      float v = acc[r] + red[ntile * 256 + (r + 8 * grp) * 16 + nn];
      int row = q0 + r + 8 * grp;
      int col = h * DKv + ntile * 16 + nn;
      Ocat[((size_t)b * Tt + row) * Cc + col] = (bf16)v;
    }
  }
}

// ---------------------------------------------------------------- launch
extern "C" void kernel_launch(void* const* d_in, const int* in_sizes, int n_in,
                              void* d_out, int out_size, void* d_ws, size_t ws_size,
                              hipStream_t stream) {
  const float* x     = (const float*)d_in[0];
  const float* ln1_g = (const float*)d_in[1];
  const float* ln1_b = (const float*)d_in[2];
  const float* w_qkv = (const float*)d_in[3];
  const float* a_hd  = (const float*)d_in[4];
  const float* w_out = (const float*)d_in[5];
  const float* ln2_g = (const float*)d_in[6];
  const float* ln2_b = (const float*)d_in[7];
  const float* w1    = (const float*)d_in[8];
  const float* b1    = (const float*)d_in[9];
  const float* w2    = (const float*)d_in[10];
  const float* b2    = (const float*)d_in[11];
  const float* alpha = (const float*)d_in[12];
  float* out = (float*)d_out;

  char* p = (char*)d_ws;
  auto alloc = [&](size_t bytes) -> char* {
    char* r = p; p += (bytes + 255) & ~(size_t)255; return r;
  };
  bf16* wqkvT = (bf16*)alloc((size_t)3 * Cc * Cc * 2);
  bf16* woutT = (bf16*)alloc((size_t)Cc * Cc * 2);
  bf16* w1T   = (bf16*)alloc((size_t)HID * Cc * 2);
  bf16* w2T   = (bf16*)alloc((size_t)Cc * HID * 2);
  bf16* xn1   = (bf16*)alloc((size_t)Mrows * Cc * 2);
  bf16* qkv   = (bf16*)alloc((size_t)Mrows * 3 * Cc * 2);
  bf16* Qb    = (bf16*)alloc((size_t)Bsz * Hh * Tt * DKv * 2);
  bf16* Kb    = (bf16*)alloc((size_t)Bsz * Hh * Tt * DKv * 2);
  bf16* Vtb   = (bf16*)alloc((size_t)Bsz * Hh * DKv * Tt * 2);
  bf16* acat  = (bf16*)alloc((size_t)Mrows * Cc * 2);
  float* aprj = (float*)alloc((size_t)Mrows * Cc * 4);
  float* x1   = (float*)alloc((size_t)Mrows * Cc * 4);
  bf16* xn2   = (bf16*)alloc((size_t)Mrows * Cc * 2);
  bf16* hbuf  = (bf16*)alloc((size_t)Mrows * HID * 2);
  float* ffn  = (float*)alloc((size_t)Mrows * Cc * 4);

  auto cdiv = [](size_t a, size_t b) { return (unsigned)((a + b - 1) / b); };

  // weights -> bf16, transposed to [N][K]
  cvt_transpose_kernel<<<cdiv((size_t)Cc * 3 * Cc, 256), 256, 0, stream>>>(w_qkv, wqkvT, Cc, 3 * Cc);
  cvt_transpose_kernel<<<cdiv((size_t)Cc * Cc, 256), 256, 0, stream>>>(w_out, woutT, Cc, Cc);
  cvt_transpose_kernel<<<cdiv((size_t)Cc * HID, 256), 256, 0, stream>>>(w1, w1T, Cc, HID);
  cvt_transpose_kernel<<<cdiv((size_t)HID * Cc, 256), 256, 0, stream>>>(w2, w2T, HID, Cc);

  // LN1 -> xn1
  layernorm_kernel<<<Mrows, 256, 0, stream>>>(x, ln1_g, ln1_b, xn1);

  // QKV projection
  gemm_bf16_kernel<true, false><<<dim3(3 * Cc / 128, Mrows / 128), 256, 0, stream>>>(
      xn1, wqkvT, qkv, nullptr, Mrows, 3 * Cc, Cc);

  // split heads (scale folded into Q), V transposed
  split_qkv_kernel<<<cdiv((size_t)Mrows * 3 * Cc, 256), 256, 0, stream>>>(qkv, Qb, Kb, Vtb);

  // fused dense+sparse attention
  size_t attn_lds = (size_t)16 * Tt * 4 + (size_t)16 * Tt * 2 + 4 * 16 * 16 * 4;
  attn_kernel<<<dim3(Tt / 16, Hh, Bsz), 256, attn_lds, stream>>>(Qb, Kb, Vtb, a_hd, acat);

  // output projection + residual
  gemm_bf16_kernel<false, false><<<dim3(Cc / 128, Mrows / 128), 256, 0, stream>>>(
      acat, woutT, aprj, nullptr, Mrows, Cc, Cc);
  resid_kernel<<<cdiv((size_t)Mrows * Cc, 256), 256, 0, stream>>>(x, aprj, alpha, x1, Mrows * Cc);

  // LN2 + FFN + residual
  layernorm_kernel<<<Mrows, 256, 0, stream>>>(x1, ln2_g, ln2_b, xn2);
  gemm_bf16_kernel<true, true><<<dim3(HID / 128, Mrows / 128), 256, 0, stream>>>(
      xn2, w1T, hbuf, b1, Mrows, HID, Cc);
  gemm_bf16_kernel<false, false><<<dim3(Cc / 128, Mrows / 128), 256, 0, stream>>>(
      hbuf, w2T, ffn, b2, Mrows, Cc, HID);
  resid_kernel<<<cdiv((size_t)Mrows * Cc, 256), 256, 0, stream>>>(x1, ffn, alpha, out, Mrows * Cc);
}
